// GIN_13597866459249
// MI455X (gfx1250) — matrix operand order repaired and checked by reading, not compiled, not run on hardware
//
#include <hip/hip_runtime.h>
#include <hip/hip_bf16.h>

typedef __attribute__((ext_vector_type(16))) __bf16  v16bf;
typedef __attribute__((ext_vector_type(8)))  float   v8f;

// ---------------------------------------------------------------------------
// helpers
// ---------------------------------------------------------------------------
__device__ __forceinline__ unsigned int f32_to_bf16_rne(float f) {
  unsigned int u = __float_as_uint(f);
  unsigned int r = u + 0x7FFFu + ((u >> 16) & 1u);   // round-to-nearest-even
  return r >> 16;
}
__device__ __forceinline__ unsigned int pack_bf16(float lo, float hi) {
  return (f32_to_bf16_rne(hi) << 16) | f32_to_bf16_rne(lo);
}

union Frag32 {            // 32 bytes: two uint4 <-> one v16bf
  uint4 q[2];
  v16bf b;
};

__global__ void fill_zero_kernel(float* __restrict__ p, long n) {
  long i = (long)blockIdx.x * blockDim.x + threadIdx.x;
  long stride = (long)gridDim.x * blockDim.x;
  for (; i < n; i += stride) p[i] = 0.0f;
}

// agg[dst[e], d] += h[src[e], d];  (1<<kshift) lanes per edge
__global__ void scatter_add_kernel(const float* __restrict__ h, int ldh,
                                   const int* __restrict__ src,
                                   const int* __restrict__ dst,
                                   float* __restrict__ agg, int ldagg,
                                   int n_edges, int K, int kshift) {
  int d  = threadIdx.x & ((1 << kshift) - 1);
  int le = threadIdx.x >> kshift;
  int e  = blockIdx.x * (blockDim.x >> kshift) + le;
  if (e >= n_edges || d >= K) return;
  int s = src[e];
  int t = dst[e];
  atomicAdd(&agg[(long)t * ldagg + d], h[(long)s * ldh + d]);
}

// g[batch[n], d] += h[n, d]   (D = 64)
__global__ void pool_kernel(const float* __restrict__ h,
                            const int* __restrict__ batch,
                            float* __restrict__ g, int n_nodes) {
  int idx = blockIdx.x * blockDim.x + threadIdx.x;
  if (idx >= (n_nodes << 6)) return;
  int n = idx >> 6;
  int d = idx & 63;
  atomicAdd(&g[(long)batch[n] * 64 + d], h[idx]);
}

// ---------------------------------------------------------------------------
// WMMA GEMM:  C[:, col_ofs : col_ofs+NC] = act(bn((A [+A2]) @ W + bias))
//   A : M x K f32 (ldA), K even, K <= 192;  A2 optional addend (ldA2)
//   W : K x NC f32 row-major;  C : f32 (ldC)
// Block = 128 thr = 4 waves; block tile = 64 rows x 64 cols.
// * W panel staged ONCE into LDS in B-fragment order (<= 24 KB).
// * A tile double-buffered; next chunk staged while current 4 WMMAs run.
// * All staging loads are branchless (clamped address + zero-select).
// * Each fragment = two ds_load_b128; B fragment reused by 4 WMMAs.
// ---------------------------------------------------------------------------
template <bool HAS_A2, bool HAS_BN, bool RELU>
__global__ __launch_bounds__(128) void gemm_wmma_t(
    const float* __restrict__ A, int ldA,
    const float* __restrict__ A2, int ldA2,
    const float* __restrict__ W,
    const float* __restrict__ bias,
    const float* __restrict__ bn_g, const float* __restrict__ bn_b,
    const float* __restrict__ bn_m, const float* __restrict__ bn_v,
    float* __restrict__ C, int ldC, int col_ofs,
    int M, int K, int NC) {
  __shared__ __align__(16) unsigned int sA[2][64 * 16];  // ping-pong A tiles
  __shared__ __align__(16) unsigned int sW[6 * 1024];    // whole W panel (K<=192)

  const int tid  = threadIdx.x;
  const int wave = tid >> 5;
  const int lane = tid & 31;
  const int row0 = blockIdx.x << 6;
  const int n0   = wave << 4;
  const int lh   = lane >> 4;       // half-wave 0/1
  const int l16  = lane & 15;
  const int nchunks = (K + 31) >> 5;

  // ---- stage the full W panel once, fragment order, branchless ----
  // chunk c, uint index (c<<10) + ((n*2+h)*8 + q) holds k = c*32+16h+2q, +1
  for (int i = tid; i < (nchunks << 10); i += 128) {
    int c   = i >> 10;
    int rem = i & 1023;
    int nn  = rem >> 4;
    int h   = (rem >> 3) & 1;
    int q   = rem & 7;
    int gk  = (c << 5) + (h << 4) + (q << 1);
    int gkc = gk < K ? gk : (K - 2);
    int nnc = nn < NC ? nn : (NC - 1);
    float lo = W[(long)gkc * NC + nnc];
    float hi = W[(long)(gkc + 1) * NC + nnc];
    bool ok  = (gk < K) && (nn < NC);
    sW[(c << 10) + (((nn << 1) + h) << 3) + q] =
        pack_bf16(ok ? lo : 0.0f, ok ? hi : 0.0f);
  }

  // branchless A-tile stager (64 rows x 16 bf16-pairs)
  auto stageA = [&](int k0, int p) {
    for (int i = tid; i < 64 * 16; i += 128) {
      int r  = i >> 4;
      int gr = row0 + r;
      gr = gr < M ? gr : (M - 1);          // clamp; stores guarded later
      int gk  = k0 + ((i & 15) << 1);
      int gkc = gk < K ? gk : (K - 2);     // clamp; K even
      float2 v = *(const float2*)(A + (long)gr * ldA + gkc);
      if (HAS_A2) {
        float2 w = *(const float2*)(A2 + (long)gr * ldA2 + gkc);
        v.x += w.x; v.y += w.y;
      }
      bool ok = gk < K;
      sA[p][i] = pack_bf16(ok ? v.x : 0.0f, ok ? v.y : 0.0f);
    }
  };

  const v8f vzero = {0.f, 0.f, 0.f, 0.f, 0.f, 0.f, 0.f, 0.f};
  v8f acc[4] = {vzero, vzero, vzero, vzero};

  stageA(0, 0);
  __syncthreads();

  for (int c = 0; c < nchunks; ++c) {
    const int p = c & 1;

    // pull this chunk's fragments into registers (2 x ds_load_b128 each)
    const uint4* pW = (const uint4*)sW;
    Frag32 fb;
    const int wi = (c << 8) + (((((n0 + l16) << 1) + lh)) << 1);
    fb.q[0] = pW[wi];
    fb.q[1] = pW[wi + 1];

    const uint4* pA = (const uint4*)sA[p];
    Frag32 fa[4];
#pragma unroll
    for (int rt = 0; rt < 4; ++rt) {
      int ai = ((rt << 4) + l16) << 2;
      fa[rt].q[0] = pA[ai + (lh << 1)];
      fa[rt].q[1] = pA[ai + (lh << 1) + 1];
    }

    // stage next chunk into the other buffer while WMMAs execute
    if (c + 1 < nchunks) stageA((c + 1) << 5, p ^ 1);

#pragma unroll
    for (int rt = 0; rt < 4; ++rt)
      acc[rt] = __builtin_amdgcn_wmma_f32_16x16x32_bf16(
          false, fa[rt].b, false, fb.b, (short)0, acc[rt], false, false);

    __syncthreads();   // one barrier per chunk (ping-pong)
  }

  // ---- epilogue: D layout VGPR r -> row (r + 8*lh), col = lane%16 ----
  const int n = n0 + l16;
  if (n < NC) {
    float sc = 1.0f, sh = 0.0f;
    if (HAS_BN) {
      sc = bn_g[n] * rsqrtf(bn_v[n] + 1e-5f);
      sh = bn_b[n] - bn_m[n] * sc;
    }
    const float bb = bias[n];
#pragma unroll
    for (int rt = 0; rt < 4; ++rt) {
#pragma unroll
      for (int r = 0; r < 8; ++r) {
        int m = row0 + (rt << 4) + r + (lh << 3);
        if (m < M) {
          float v = acc[rt][r] + bb;
          if (HAS_BN) v = v * sc + sh;
          if (RELU) v = fmaxf(v, 0.0f);
          C[(long)m * ldC + col_ofs + n] = v;
        }
      }
    }
  }
}

// ---------------------------------------------------------------------------
// driver
// ---------------------------------------------------------------------------
extern "C" void kernel_launch(void* const* d_in, const int* in_sizes, int n_in,
                              void* d_out, int out_size, void* d_ws, size_t ws_size,
                              hipStream_t stream) {
  (void)in_sizes; (void)n_in; (void)out_size; (void)ws_size;

  const float* x      = (const float*)d_in[0];
  const int*   edge   = (const int*)d_in[1];
  const int*   batch  = (const int*)d_in[2];
  const float* w1[3]  = {(const float*)d_in[3],  (const float*)d_in[7],  (const float*)d_in[11]};
  const float* b1[3]  = {(const float*)d_in[4],  (const float*)d_in[8],  (const float*)d_in[12]};
  const float* w2[3]  = {(const float*)d_in[5],  (const float*)d_in[9],  (const float*)d_in[13]};
  const float* b2[3]  = {(const float*)d_in[6],  (const float*)d_in[10], (const float*)d_in[14]};
  const float* jk_w   = (const float*)d_in[15];
  const float* jk_b   = (const float*)d_in[16];
  const float* ffn_w1 = (const float*)d_in[17];
  const float* ffn_b1 = (const float*)d_in[18];
  const float* bn_g   = (const float*)d_in[19];
  const float* bn_b   = (const float*)d_in[20];
  const float* bn_m   = (const float*)d_in[21];
  const float* bn_v   = (const float*)d_in[22];
  const float* ffn_w2 = (const float*)d_in[23];
  const float* ffn_b2 = (const float*)d_in[24];
  const float* out_w  = (const float*)d_in[25];
  const float* out_b  = (const float*)d_in[26];

  const int N = 50000, E = 800000, D = 64, IN = 100, G = 512, OUT = 24;
  const int* src = edge;          // edge_index[0]
  const int* dst = edge + E;      // edge_index[1]

  // workspace (f32): agg | hcat(N x 192) | tmp(N x 64) | g | t1 | t2
  float* ws   = (float*)d_ws;
  float* agg  = ws;
  float* hcat = agg  + (size_t)N * IN;
  float* tmp  = hcat + (size_t)N * 192;
  float* gbuf = tmp  + (size_t)N * D;
  float* t1   = gbuf + (size_t)G * D;
  float* t2   = t1   + (size_t)G * D;

  const int gemmN = (N + 63) / 64;   // 782
  const int gemmG = (G + 63) / 64;   // 8

  // ---- layer 0 (input dim 100) ----
  fill_zero_kernel<<<2048, 256, 0, stream>>>(agg, (long)N * IN);
  scatter_add_kernel<<<(E + 1) / 2, 256, 0, stream>>>(x, IN, src, dst, agg, IN, E, IN, 7);
  gemm_wmma_t<true, false, true><<<gemmN, 128, 0, stream>>>(
      x, IN, agg, IN, w1[0], b1[0], nullptr, nullptr, nullptr, nullptr,
      tmp, D, 0, N, IN, D);
  gemm_wmma_t<false, false, true><<<gemmN, 128, 0, stream>>>(
      tmp, D, nullptr, 0, w2[0], b2[0], nullptr, nullptr, nullptr, nullptr,
      hcat, 192, 0, N, D, D);

  // ---- layers 1..2 ----
  for (int l = 1; l < 3; ++l) {
    const float* h = hcat + (size_t)(l - 1) * 64;   // previous layer's columns
    fill_zero_kernel<<<2048, 256, 0, stream>>>(agg, (long)N * D);
    scatter_add_kernel<<<(E + 3) / 4, 256, 0, stream>>>(h, 192, src, dst, agg, D, E, D, 6);
    gemm_wmma_t<true, false, true><<<gemmN, 128, 0, stream>>>(
        h, 192, agg, D, w1[l], b1[l], nullptr, nullptr, nullptr, nullptr,
        tmp, D, 0, N, D, D);
    gemm_wmma_t<false, false, true><<<gemmN, 128, 0, stream>>>(
        tmp, D, nullptr, 0, w2[l], b2[l], nullptr, nullptr, nullptr, nullptr,
        hcat, 192, l * 64, N, D, D);
  }

  // ---- JK 'cat' linear: hcat (N x 192) @ jk_w -> tmp (N x 64), no relu
  gemm_wmma_t<false, false, false><<<gemmN, 128, 0, stream>>>(
      hcat, 192, nullptr, 0, jk_w, jk_b, nullptr, nullptr, nullptr, nullptr,
      tmp, D, 0, N, 192, D);

  // ---- global_add_pool -> gbuf (512 x 64)
  fill_zero_kernel<<<128, 256, 0, stream>>>(gbuf, (long)G * D);
  pool_kernel<<<(N * 64 + 255) / 256, 256, 0, stream>>>(tmp, batch, gbuf, N);

  // ---- head: lin -> BN(eval) -> relu | lin | out linear (512 x 24)
  gemm_wmma_t<false, true, true><<<gemmG, 128, 0, stream>>>(
      gbuf, D, nullptr, 0, ffn_w1, ffn_b1, bn_g, bn_b, bn_m, bn_v,
      t1, D, 0, G, D, D);
  gemm_wmma_t<false, false, false><<<gemmG, 128, 0, stream>>>(
      t1, D, nullptr, 0, ffn_w2, ffn_b2, nullptr, nullptr, nullptr, nullptr,
      t2, D, 0, G, D, D);
  gemm_wmma_t<false, false, false><<<gemmG, 128, 0, stream>>>(
      t2, D, nullptr, 0, out_w, out_b, nullptr, nullptr, nullptr, nullptr,
      (float*)d_out, OUT, 0, G, D, OUT);
}
